// DeepHamActor_43327630082672
// MI455X (gfx1250) — matrix operand order repaired and checked by reading, not compiled, not run on hardware
//
#include <hip/hip_runtime.h>
#include <hip/hip_bf16.h>

#define DIN  128
#define DMID 512
#define DHID 256
#define LRELU_ALPHA 0.1f

typedef __attribute__((ext_vector_type(16))) __bf16 v16bf;
typedef __attribute__((ext_vector_type(8)))  float  v8f;

__device__ __forceinline__ unsigned short f2bf(float x) {
  unsigned u = __float_as_uint(x);
  u += 0x7FFFu + ((u >> 16) & 1u);          // round-to-nearest-even
  return (unsigned short)(u >> 16);
}

// Native gfx1250 no-return f32 atomic add (GLOBAL_ATOMIC_ADD_F32, STOREcnt).
// Avoids a possible CAS-loop expansion of atomicrmw fadd.
__device__ __forceinline__ void atomic_add_f32(float* p, float v) {
  asm volatile("global_atomic_add_f32 %0, %1, off" : : "v"(p), "v"(v) : "memory");
}

// ---------------- degree / normalization ----------------
__global__ void k_fill1(float* x, int n) {
  int i = blockIdx.x * blockDim.x + threadIdx.x;
  if (i < n) x[i] = 1.0f;                   // self-loop contribution
}
__global__ void k_deg_count(const int* __restrict__ dst, float* deg, int e) {
  int i = blockIdx.x * blockDim.x + threadIdx.x;
  if (i < e) atomic_add_f32(&deg[dst[i]], 1.0f);
}
__global__ void k_rsqrt_inplace(float* x, int n) {
  int i = blockIdx.x * blockDim.x + threadIdx.x;
  if (i < n) x[i] = rsqrtf(fmaxf(x[i], 1.0f));
}

// ---------------- fp32 -> bf16 conversions ----------------
__global__ void k_tobf16(const float* __restrict__ x, unsigned short* __restrict__ y,
                         unsigned total) {
  unsigned i = blockIdx.x * blockDim.x + threadIdx.x;
  if (i < total) y[i] = f2bf(x[i]);
}
// W: [K x M] row-major fp32  ->  Wt: [M x K] row-major bf16 (i.e. B^T)
__global__ void k_wtrans_bf16(const float* __restrict__ W, unsigned short* __restrict__ Wt,
                              int K, int M) {
  unsigned i = blockIdx.x * blockDim.x + threadIdx.x;
  if (i >= (unsigned)K * M) return;
  int k = i / M, m = i - k * M;
  Wt[(size_t)m * K + k] = f2bf(W[i]);
}

// ---------------- WMMA bf16 GEMM: C[MxN] = A[MxK] * Bt[NxK]^T ----------------
// 8 waves/block, each wave = one 16x16 tile; block tile = 16 rows x 128 cols.
// Requires: rows % 16 == 0, K % 32 == 0, Ncols % 128 == 0 (all true here).
__global__ __launch_bounds__(256) void k_gemm_bf16(
    const unsigned short* __restrict__ A,   // [rows x K] bf16 row-major
    const unsigned short* __restrict__ Bt,  // [Ncols x K] bf16 row-major
    float* __restrict__ C,                  // [rows x Ncols] fp32
    int K, int Ncols) {
  const int lane = threadIdx.x & 31;
  const int wave = threadIdx.x >> 5;
  const int row0 = blockIdx.x << 4;
  const int col0 = (blockIdx.y << 7) + (wave << 4);
  const int l16  = lane & 15;
  const int hi   = lane >> 4;     // lane group 0/1
  const int aoff = hi << 3;       // A frag: hi lanes hold K+8 (ISA 16-bit A 16x32 layout)
  const int boff = hi << 3;       // B frag: hi lanes hold K+8 (same k order as A)

  const unsigned short* arow = A  + (size_t)(row0 + l16) * K;
  const unsigned short* brow = Bt + (size_t)(col0 + l16) * K;

  union Frag { v16bf v; unsigned u[8]; };
  v8f acc = {0.f, 0.f, 0.f, 0.f, 0.f, 0.f, 0.f, 0.f};

  for (int k0 = 0; k0 < K; k0 += 32) {
    Frag a, b;
#pragma unroll
    for (int r = 0; r < 8; ++r) {
      const int ka = ((r < 4) ? (r << 1) : (16 + ((r - 4) << 1))) + aoff;
      const int kb = ((r < 4) ? (r << 1) : (16 + ((r - 4) << 1))) + boff;
      a.u[r] = *(const unsigned*)(arow + k0 + ka);   // 2 consecutive-K bf16
      b.u[r] = *(const unsigned*)(brow + k0 + kb);
    }
    acc = __builtin_amdgcn_wmma_f32_16x16x32_bf16(
        false, a.v, false, b.v, (short)0, acc, false, false);
  }

  float* crow = C + (size_t)row0 * Ncols + col0;
  const int mbase = hi << 3;      // VGPR r: M = r (lanes 0-15) / 8+r (lanes 16-31)
#pragma unroll
  for (int r = 0; r < 8; ++r)
    crow[(size_t)(r + mbase) * Ncols + l16] = acc[r];
}

// ---------------- GCN aggregation ----------------
// out[n,f] = dis[n]^2 * H[n,f] + b[f]   (self-loop + bias, also initializes out)
__global__ void k_self_bias(const float* __restrict__ H, const float* __restrict__ dis,
                            const float* __restrict__ b, float* __restrict__ out,
                            int cols, unsigned total) {
  unsigned i = blockIdx.x * blockDim.x + threadIdx.x;
  if (i >= total) return;
  int node = i / cols, f = i - node * cols;
  float d = dis[node];
  out[i] = d * d * H[i] + b[f];
}
// out[dst] += dis[src]*dis[dst] * H[src]  (float4 per thread, coalesced per edge)
__global__ void k_edge_scatter(const int* __restrict__ src, const int* __restrict__ dst,
                               const float* __restrict__ dis, const float* __restrict__ H,
                               float* __restrict__ out, int e, int cols) {
  const unsigned tid = blockIdx.x * blockDim.x + threadIdx.x;
  const int chunks = cols >> 2;
  if (tid >= (unsigned)e * (unsigned)chunks) return;
  const int ei = tid / chunks;
  const int c  = tid - ei * chunks;
  const int s = src[ei], d = dst[ei];
  const float nrm = dis[s] * dis[d];
  const float4 h = *(const float4*)(H + (size_t)s * cols + (c << 2));
  float* o = out + (size_t)d * cols + (c << 2);
  atomic_add_f32(o + 0, nrm * h.x);
  atomic_add_f32(o + 1, nrm * h.y);
  atomic_add_f32(o + 2, nrm * h.z);
  atomic_add_f32(o + 3, nrm * h.w);
}
__global__ void k_tanh_inplace(float* x, unsigned total) {
  unsigned i = blockIdx.x * blockDim.x + threadIdx.x;
  if (i < total) x[i] = tanhf(x[i]);
}

// ---------------- MLP head: s[n] = sum_h lrelu(S1[n,h]+bm1[h]) * Wm2[h] + bm2 ----------------
__global__ __launch_bounds__(256) void k_mlp_head(
    const float* __restrict__ S1, const float* __restrict__ bm1,
    const float* __restrict__ Wm2, const float* __restrict__ bm2,
    float* __restrict__ s) {
  const int node = blockIdx.x;
  const int t = threadIdx.x;                 // 256 == DHID
  float v = S1[(size_t)node * DHID + t] + bm1[t];
  v = (v > 0.f) ? v : LRELU_ALPHA * v;
  v *= Wm2[t];
  for (int o = 16; o; o >>= 1) v += __shfl_down(v, o, 32);
  __shared__ float red[8];
  if ((t & 31) == 0) red[t >> 5] = v;
  __syncthreads();
  if (t == 0) {
    float a = red[0];
#pragma unroll
    for (int i = 1; i < 8; ++i) a += red[i];
    s[node] = a + bm2[0];
  }
}

// ---------------- single-block softmax over n scores ----------------
__global__ __launch_bounds__(1024) void k_softmax(const float* __restrict__ s,
                                                  float* __restrict__ out, int n) {
  __shared__ float red[32];
  __shared__ float bc[2];
  const int t = threadIdx.x, lane = t & 31, wv = t >> 5;
  float m = -3.4e38f;
  for (int i = t; i < n; i += 1024) m = fmaxf(m, s[i]);
  for (int o = 16; o; o >>= 1) m = fmaxf(m, __shfl_down(m, o, 32));
  if (lane == 0) red[wv] = m;
  __syncthreads();
  if (wv == 0) {
    float mm = red[lane];
    for (int o = 16; o; o >>= 1) mm = fmaxf(mm, __shfl_down(mm, o, 32));
    if (lane == 0) bc[0] = mm;
  }
  __syncthreads();
  m = bc[0];
  float sum = 0.f;
  for (int i = t; i < n; i += 1024) sum += __expf(s[i] - m);
  for (int o = 16; o; o >>= 1) sum += __shfl_down(sum, o, 32);
  if (lane == 0) red[wv] = sum;
  __syncthreads();
  if (wv == 0) {
    float ss = red[lane];
    for (int o = 16; o; o >>= 1) ss += __shfl_down(ss, o, 32);
    if (lane == 0) bc[1] = ss;
  }
  __syncthreads();
  const float inv = 1.0f / bc[1];
  for (int i = t; i < n; i += 1024) out[i] = __expf(s[i] - m) * inv;
}

// ---------------- host driver ----------------
extern "C" void kernel_launch(void* const* d_in, const int* in_sizes, int n_in,
                              void* d_out, int out_size, void* d_ws, size_t ws_size,
                              hipStream_t stream) {
  (void)n_in; (void)out_size; (void)ws_size;
  const float* vertices = (const float*)d_in[0];
  const int*   eidx     = (const int*)d_in[1];
  const float* W1  = (const float*)d_in[3];  const float* b1  = (const float*)d_in[4];
  const float* W2  = (const float*)d_in[5];  const float* b2  = (const float*)d_in[6];
  const float* W3  = (const float*)d_in[7];  const float* b3  = (const float*)d_in[8];
  const float* Wm1 = (const float*)d_in[9];  const float* bm1 = (const float*)d_in[10];
  const float* Wm2 = (const float*)d_in[11]; const float* bm2 = (const float*)d_in[12];

  const int N = in_sizes[0] / DIN;   // 50000 (divisible by 16)
  const int E = in_sizes[1] / 2;     // 800000
  const int* esrc = eidx;
  const int* edst = eidx + E;

  char* p = (char*)d_ws;
  auto alloc = [&](size_t bytes) -> void* {
    void* r = p; p += (bytes + 255) & ~(size_t)255; return r;
  };
  float*          dis  = (float*)alloc((size_t)N * 4);
  unsigned short* W1t  = (unsigned short*)alloc((size_t)DIN  * DMID * 2);
  unsigned short* W2t  = (unsigned short*)alloc((size_t)DMID * DMID * 2);
  unsigned short* W3t  = (unsigned short*)alloc((size_t)DMID * DMID * 2);
  unsigned short* Wm1t = (unsigned short*)alloc((size_t)DMID * DHID * 2);
  unsigned short* xb   = (unsigned short*)alloc((size_t)N * DMID * 2);
  float*          Hm   = (float*)alloc((size_t)N * DMID * 4);
  float*          act  = (float*)alloc((size_t)N * DMID * 4);
  float*          sv   = (float*)alloc((size_t)N * 4);

  const dim3 b256(256);
  auto cdiv = [](long long a, long long b) { return (unsigned)((a + b - 1) / b); };

  // degree -> dis = rsqrt(deg)
  k_fill1<<<cdiv(N, 256), b256, 0, stream>>>(dis, N);
  k_deg_count<<<cdiv(E, 256), b256, 0, stream>>>(edst, dis, E);
  k_rsqrt_inplace<<<cdiv(N, 256), b256, 0, stream>>>(dis, N);

  // weights -> transposed bf16
  k_wtrans_bf16<<<cdiv((long long)DIN  * DMID, 256), b256, 0, stream>>>(W1,  W1t,  DIN,  DMID);
  k_wtrans_bf16<<<cdiv((long long)DMID * DMID, 256), b256, 0, stream>>>(W2,  W2t,  DMID, DMID);
  k_wtrans_bf16<<<cdiv((long long)DMID * DMID, 256), b256, 0, stream>>>(W3,  W3t,  DMID, DMID);
  k_wtrans_bf16<<<cdiv((long long)DMID * DHID, 256), b256, 0, stream>>>(Wm1, Wm1t, DMID, DHID);

  const unsigned totD = (unsigned)N * DMID;
  const dim3 gemmG(N / 16, DMID / 128);

  // ---- layer 1: x = vertices [N x 128] ----
  k_tobf16<<<cdiv((long long)N * DIN, 256), b256, 0, stream>>>(vertices, xb, (unsigned)N * DIN);
  k_gemm_bf16<<<gemmG, b256, 0, stream>>>(xb, W1t, Hm, DIN, DMID);
  k_self_bias<<<cdiv(totD, 256), b256, 0, stream>>>(Hm, dis, b1, act, DMID, totD);
  k_edge_scatter<<<cdiv((long long)E * (DMID / 4), 256), b256, 0, stream>>>(esrc, edst, dis, Hm, act, E, DMID);
  k_tanh_inplace<<<cdiv(totD, 256), b256, 0, stream>>>(act, totD);

  // ---- layers 2 & 3: act [N x 512] ping-pongs through xb/Hm ----
  const unsigned short* Wts[2] = { W2t, W3t };
  const float* bs[2] = { b2, b3 };
  for (int L = 0; L < 2; ++L) {
    k_tobf16<<<cdiv(totD, 256), b256, 0, stream>>>(act, xb, totD);
    k_gemm_bf16<<<gemmG, b256, 0, stream>>>(xb, Wts[L], Hm, DMID, DMID);
    k_self_bias<<<cdiv(totD, 256), b256, 0, stream>>>(Hm, dis, bs[L], act, DMID, totD);
    k_edge_scatter<<<cdiv((long long)E * (DMID / 4), 256), b256, 0, stream>>>(esrc, edst, dis, Hm, act, E, DMID);
    k_tanh_inplace<<<cdiv(totD, 256), b256, 0, stream>>>(act, totD);
  }

  // ---- MLP head + softmax ----
  k_tobf16<<<cdiv(totD, 256), b256, 0, stream>>>(act, xb, totD);
  k_gemm_bf16<<<dim3(N / 16, DHID / 128), b256, 0, stream>>>(xb, Wm1t, Hm, DMID, DHID);
  k_mlp_head<<<N, b256, 0, stream>>>(Hm, bm1, Wm2, bm2, sv);
  k_softmax<<<1, 1024, 0, stream>>>(sv, (float*)d_out, N);
}
